// TwoStream_Core_60653528154521
// MI455X (gfx1250) — compile-verified, matrix operand order checked by reading, not run on hardware
//
#include <hip/hip_runtime.h>
#include <hip/hip_bf16.h>
#include <math.h>

typedef __attribute__((ext_vector_type(16))) __bf16 v16bf;
typedef __attribute__((ext_vector_type(8)))  float  v8f;

#define BM 128
#define BN 128
#define BK 64
#define TPB 256
#define LDSS 72   // padded row stride in bf16 elements (144B: 16B aligned, conflict-free)

static __device__ __forceinline__ unsigned short f2bf(float f) {
    union { float f; unsigned u; } v; v.f = f;
    unsigned r = v.u + 0x7FFFu + ((v.u >> 16) & 1u);   // round-to-nearest-even
    return (unsigned short)(r >> 16);
}

union ABFrag { uint4 q[2]; v16bf v; };

// C = A @ B^T (+bias), A fp32 row-major with lda==1024 per 1024-wide K-region
// (3 region pointers -> fused [x | y | h] concat), B fp32 row-major [N,K]
// split at K1 between B0 (ldb0) and B1 (ldb1).
// EPI==0: store C+bias to Cout.  EPI==1: attention score epilogue.
template<int EPI>
__global__ __launch_bounds__(TPB)
void gemm_core(const float* __restrict__ A0, const float* __restrict__ A1,
               const float* __restrict__ A2,
               const float* __restrict__ B0, const float* __restrict__ B1,
               int K1, int ldb0, int ldb1,
               int M, int N, int K,
               const float* __restrict__ bias0, const float* __restrict__ bias1,
               float* __restrict__ Cout, int ldc,
               const float* __restrict__ atth, const float* __restrict__ w_alpha,
               float* __restrict__ scores)
{
    __shared__ __align__(16) unsigned short As[BM * LDSS];
    __shared__ __align__(16) unsigned short Bs[BN * LDSS];

    const int tid  = threadIdx.x;
    const int lane = tid & 31;
    const int wv   = tid >> 5;
    const int wm   = wv >> 2;       // 0..1  (64 rows of C each)
    const int wn   = wv & 3;        // 0..3  (32 cols of C each)
    const int half = lane >> 4;     // k-group select per WMMA layout
    const int rl   = lane & 15;

    const int bm = blockIdx.y * BM;
    const int bn = blockIdx.x * BN;

    v8f zero = {};
    v8f acc[4][2];
    #pragma unroll
    for (int i = 0; i < 4; ++i)
        #pragma unroll
        for (int j = 0; j < 2; ++j) acc[i][j] = zero;

    const int r0 = tid & 15;        // float4 column within 64-wide k tile
    const int rr = tid >> 4;        // row base, 16 rows per pass

    for (int k0 = 0; k0 < K; k0 += BK) {
        // -------- stage global -> registers (fp32)
        float4 ar[8], br[8];
        {
            const int reg = k0 >> 10;
            const float* Ap = (reg == 0) ? A0 : ((reg == 1) ? A1 : A2);
            const int ac = (k0 & 1023) + r0 * 4;
            #pragma unroll
            for (int p = 0; p < 8; ++p) {
                int row = rr + p * 16;
                ar[p] = *(const float4*)(Ap + (size_t)(bm + row) * 1024 + ac);
            }
            const float* Bp; int bc, ldb;
            if (k0 < K1) { Bp = B0; bc = k0;      ldb = ldb0; }
            else         { Bp = B1; bc = k0 - K1; ldb = ldb1; }
            bc += r0 * 4;
            #pragma unroll
            for (int p = 0; p < 8; ++p) {
                int row = rr + p * 16;
                br[p] = *(const float4*)(Bp + (size_t)(bn + row) * ldb + bc);
            }
        }
        __syncthreads();
        // -------- registers -> LDS as packed bf16 (ds_store_b64)
        #pragma unroll
        for (int p = 0; p < 8; ++p) {
            int row = rr + p * 16;
            uint2 pa, pb;
            pa.x = (unsigned)f2bf(ar[p].x) | ((unsigned)f2bf(ar[p].y) << 16);
            pa.y = (unsigned)f2bf(ar[p].z) | ((unsigned)f2bf(ar[p].w) << 16);
            pb.x = (unsigned)f2bf(br[p].x) | ((unsigned)f2bf(br[p].y) << 16);
            pb.y = (unsigned)f2bf(br[p].z) | ((unsigned)f2bf(br[p].w) << 16);
            *(uint2*)&As[row * LDSS + r0 * 4] = pa;
            *(uint2*)&Bs[row * LDSS + r0 * 4] = pb;
        }
        __syncthreads();
        // -------- two K=32 WMMA steps per tile
        #pragma unroll
        for (int kk = 0; kk < 2; ++kk) {
            const int kb = kk * 32 + half * 8;
            ABFrag af[4], bfr[2];
            #pragma unroll
            for (int am = 0; am < 4; ++am) {
                int row = wm * 64 + am * 16 + rl;
                af[am].q[0] = *(const uint4*)&As[row * LDSS + kb];
                af[am].q[1] = *(const uint4*)&As[row * LDSS + kb + 16];
            }
            #pragma unroll
            for (int bq = 0; bq < 2; ++bq) {
                int row = wn * 32 + bq * 16 + rl;
                bfr[bq].q[0] = *(const uint4*)&Bs[row * LDSS + kb];
                bfr[bq].q[1] = *(const uint4*)&Bs[row * LDSS + kb + 16];
            }
            #pragma unroll
            for (int am = 0; am < 4; ++am)
                #pragma unroll
                for (int bq = 0; bq < 2; ++bq)
                    acc[am][bq] = __builtin_amdgcn_wmma_f32_16x16x32_bf16(
                        false, af[am].v, false, bfr[bq].v,
                        (short)0, acc[am][bq], false, false);
        }
    }

    if (EPI == 0) {
        #pragma unroll
        for (int bq = 0; bq < 2; ++bq) {
            int n = bn + wn * 32 + bq * 16 + rl;
            float bv = 0.f;
            if (bias0) bv += bias0[n];
            if (bias1) bv += bias1[n];
            #pragma unroll
            for (int am = 0; am < 4; ++am) {
                int mbase = bm + wm * 64 + am * 16 + half * 8;
                #pragma unroll
                for (int i = 0; i < 8; ++i)
                    Cout[(size_t)(mbase + i) * ldc + n] = acc[am][bq][i] + bv;
            }
        }
    } else {
        // scores[m] += sum_n w_alpha[n] * tanh(acc + b_ctx[n] + att_h[m/64][n])
        #pragma unroll
        for (int am = 0; am < 4; ++am) {
            int mbase = bm + wm * 64 + am * 16 + half * 8;
            #pragma unroll
            for (int i = 0; i < 8; ++i) {
                int m = mbase + i;
                int bidx = m >> 6;            // batch index (S == 64)
                float part = 0.f;
                #pragma unroll
                for (int bq = 0; bq < 2; ++bq) {
                    int n = bn + wn * 32 + bq * 16 + rl;
                    float val = acc[am][bq][i] + bias0[n]
                              + atth[(size_t)bidx * 512 + n];
                    part += w_alpha[n] * tanhf(val);
                }
                part += __shfl_xor(part, 1, 32);
                part += __shfl_xor(part, 2, 32);
                part += __shfl_xor(part, 4, 32);
                part += __shfl_xor(part, 8, 32);
                if (rl == 0) atomicAdd(&scores[m], part);
            }
        }
    }
}

__global__ void zero_kernel(float* p, int n) {
    int i = blockIdx.x * blockDim.x + threadIdx.x;
    if (i < n) p[i] = 0.f;
}

__global__ void softmax_mask_kernel(const float* __restrict__ scores,
                                    const float* __restrict__ b_alpha,
                                    const int* __restrict__ mask,
                                    float* __restrict__ weight)
{
    int b = blockIdx.x;
    int t = threadIdx.x;                       // 32 threads, 2 elems each
    float s0 = scores[b * 64 + t]      + b_alpha[0];
    float s1 = scores[b * 64 + 32 + t] + b_alpha[0];
    float mx = fmaxf(s0, s1);
    for (int o = 1; o < 32; o <<= 1) mx = fmaxf(mx, __shfl_xor(mx, o, 32));
    float e0 = __expf(s0 - mx), e1 = __expf(s1 - mx);
    float sm = e0 + e1;
    for (int o = 1; o < 32; o <<= 1) sm += __shfl_xor(sm, o, 32);
    float w0 = e0 / sm, w1 = e1 / sm;
    w0 *= (float)mask[b * 64 + t];
    w1 *= (float)mask[b * 64 + 32 + t];
    float s2 = w0 + w1;
    for (int o = 1; o < 32; o <<= 1) s2 += __shfl_xor(s2, o, 32);
    weight[b * 64 + t]      = w0 / s2;
    weight[b * 64 + 32 + t] = w1 / s2;
}

__global__ __launch_bounds__(256)
void weighted_sum_kernel(const float* __restrict__ weight,
                         const float* __restrict__ clip,
                         float* __restrict__ att)
{
    __shared__ float wbuf[64];
    int b = blockIdx.x, t = threadIdx.x;
    if (t < 64) wbuf[t] = weight[b * 64 + t];
    __syncthreads();
    float4 acc = make_float4(0.f, 0.f, 0.f, 0.f);
    const float4* cp = (const float4*)(clip + (size_t)b * 64 * 1024) + t;
    for (int s = 0; s < 64; ++s) {
        float w = wbuf[s];
        float4 c = cp[s * 256];
        acc.x += w * c.x; acc.y += w * c.y; acc.z += w * c.z; acc.w += w * c.w;
    }
    ((float4*)(att + (size_t)b * 1024))[t] = acc;
}

__global__ __launch_bounds__(256)
void lstm_ew_kernel(const float* __restrict__ gates,
                    const float* __restrict__ c_prev,
                    float* __restrict__ dout, int st)
{
    int idx = blockIdx.x * 256 + threadIdx.x;  // b*1024 + r
    int b = idx >> 10, r = idx & 1023;
    const float* g = gates + (size_t)b * 4096;
    float i  = 1.f / (1.f + __expf(-g[r]));
    float f  = 1.f / (1.f + __expf(-g[r + 1024]));
    float gt = tanhf(g[r + 2048]);
    float o  = 1.f / (1.f + __expf(-g[r + 3072]));
    float cn = f * c_prev[idx] + i * gt;
    float hn = o * tanhf(cn);
    dout[(size_t)b * 2048 + st * 1024 + r]       = hn;  // output [B,2R]
    dout[2097152 + (size_t)st * 1048576 + idx]   = hn;  // new_h [2,B,R]
    dout[4194304 + (size_t)st * 1048576 + idx]   = cn;  // new_c [2,B,R]
}

extern "C" void kernel_launch(void* const* d_in, const int* in_sizes, int n_in,
                              void* d_out, int out_size, void* d_ws, size_t ws_size,
                              hipStream_t stream)
{
    (void)in_sizes; (void)n_in; (void)out_size; (void)ws_size;
    const float* xt      = (const float*)d_in[0];
    const float* event   = (const float*)d_in[2];
    const float* clip    = (const float*)d_in[3];
    const int*   mask    = (const int*)d_in[4];
    const float* state_h = (const float*)d_in[5];
    const float* state_c = (const float*)d_in[6];
    const float* W_ih0 = (const float*)d_in[7];
    const float* b_ih0 = (const float*)d_in[8];
    const float* W_hh0 = (const float*)d_in[9];
    const float* b_hh0 = (const float*)d_in[10];
    const float* W_ih1 = (const float*)d_in[11];
    const float* b_ih1 = (const float*)d_in[12];
    const float* W_hh1 = (const float*)d_in[13];
    const float* b_hh1 = (const float*)d_in[14];
    const float* W_ctx = (const float*)d_in[15];
    const float* b_ctx = (const float*)d_in[16];
    const float* W_h2a = (const float*)d_in[17];
    const float* b_h2a = (const float*)d_in[18];
    const float* W_alpha = (const float*)d_in[19];
    const float* b_alpha = (const float*)d_in[20];

    float* out = (float*)d_out;
    float* ws  = (float*)d_ws;
    float* gates  = ws;                 // 1024*4096
    float* atth   = ws + 4194304;       // 1024*512
    float* scores = ws + 4718592;       // 65536
    float* weight = ws + 4784128;       // 65536
    float* att    = ws + 4849664;       // 1024*1024

    const float* h0 = state_h;
    const float* h1 = state_h + 1048576;
    const float* c0 = state_c;
    const float* c1 = state_c + 1048576;

    dim3 blk(TPB);

    // stream 0: gates0 = [xt|event|h0] @ [W_ih0|W_hh0]^T + b   (K fused = 3072)
    gemm_core<0><<<dim3(32, 8), blk, 0, stream>>>(
        xt, event, h0, W_ih0, W_hh0, 2048, 2048, 1024,
        1024, 4096, 3072, b_ih0, b_hh0, gates, 4096,
        nullptr, nullptr, nullptr);
    lstm_ew_kernel<<<dim3(4096), blk, 0, stream>>>(gates, c0, out, 0);

    // att_h = pre_h1 @ W_h2a^T + b_h2a
    gemm_core<0><<<dim3(4, 8), blk, 0, stream>>>(
        h1, h1, h1, W_h2a, W_h2a, 1024, 1024, 1024,
        1024, 512, 1024, b_h2a, nullptr, atth, 512,
        nullptr, nullptr, nullptr);

    // scores = W_alpha . tanh(clip @ W_ctx^T + b_ctx + att_h)  (fused epilogue)
    zero_kernel<<<dim3(256), blk, 0, stream>>>(scores, 65536);
    gemm_core<1><<<dim3(4, 512), blk, 0, stream>>>(
        clip, clip, clip, W_ctx, W_ctx, 1024, 1024, 1024,
        65536, 512, 1024, b_ctx, nullptr, nullptr, 512,
        atth, W_alpha, scores);

    softmax_mask_kernel<<<dim3(1024), dim3(32), 0, stream>>>(scores, b_alpha, mask, weight);
    weighted_sum_kernel<<<dim3(1024), blk, 0, stream>>>(weight, clip, att);

    // stream 1: gates1 = [xt|att|h1] @ [W_ih1|W_hh1]^T + b
    gemm_core<0><<<dim3(32, 8), blk, 0, stream>>>(
        xt, att, h1, W_ih1, W_hh1, 2048, 2048, 1024,
        1024, 4096, 3072, b_ih1, b_hh1, gates, 4096,
        nullptr, nullptr, nullptr);
    lstm_ew_kernel<<<dim3(4096), blk, 0, stream>>>(gates, c1, out, 1);
}
